// RelationAwareGAT_49289044689185
// MI455X (gfx1250) — matrix operand order repaired and checked by reading, not compiled, not run on hardware
//
#include <hip/hip_runtime.h>
#include <math.h>

#define Bdim 8
#define Nn   1024
#define Dd   768
#define DFf  1536
#define Rr   10
#define Hh   12
#define DHh  64
#define Mrows (Bdim*Nn)

typedef unsigned short u16;
typedef __attribute__((ext_vector_type(16))) __bf16 bf16x16;
typedef __attribute__((ext_vector_type(8)))  float  f32x8;
typedef __attribute__((ext_vector_type(4)))  unsigned int v4u;
typedef __attribute__((ext_vector_type(8)))  int v8i;
typedef __attribute__((ext_vector_type(4)))  int v4i;

#if __has_builtin(__builtin_amdgcn_tensor_load_to_lds)
#define HAVE_TDM 1
#else
#define HAVE_TDM 0
#endif

union FragAB { uint4 u[2]; bf16x16 v; };
union FragW  { unsigned int w[8]; bf16x16 v; };

__device__ inline u16 f2bf(float f) {
  unsigned u = __float_as_uint(f);
  unsigned r = u + 0x7fffu + ((u >> 16) & 1u);   // round-to-nearest-even
  return (u16)(r >> 16);
}

__device__ inline float redmax16(float v) {
  v = fmaxf(v, __shfl_xor(v, 1, 16));
  v = fmaxf(v, __shfl_xor(v, 2, 16));
  v = fmaxf(v, __shfl_xor(v, 4, 16));
  v = fmaxf(v, __shfl_xor(v, 8, 16));
  return v;
}
__device__ inline float redsum16(float v) {
  v += __shfl_xor(v, 1, 16);
  v += __shfl_xor(v, 2, 16);
  v += __shfl_xor(v, 4, 16);
  v += __shfl_xor(v, 8, 16);
  return v;
}

#if HAVE_TDM
// Issue a TDM 2D tile load: tile1 rows x tile0 elems of 2-byte data,
// global row stride = stride0 elems, LDS pad of 4 DWORDs per 16 DWORDs
// stored (=> 80B LDS row pitch for a 64B row).
// D# packing per CDNA5 ISA.
// This toolchain declares the 6-arg builtin:
//   (uint32x4 g0, int32x8 g1, int32x4 g2, int32x4 g3, int32x8 extra, i32 cpol)
__device__ inline void tdm_load_2d_bf16(unsigned lds_addr, const void* gptr,
                                        unsigned tile0, unsigned tile1,
                                        unsigned long long stride0) {
  unsigned long long ga = (unsigned long long)gptr;
  v4u g0;
  g0[0] = 1u;                                   // count=1 valid user descriptor
  g0[1] = lds_addr;                             // lds_addr [63:32]
  g0[2] = (unsigned)(ga & 0xffffffffu);         // global_addr [95:64]
  g0[3] = (unsigned)((ga >> 32) & 0x01ffffffu)  // global_addr [120:96]
          | (2u << 30);                         // type=2 ("image")
  v8i g1;
  unsigned w0 = (1u << 16)                      // data_size=1 (2 bytes)
              | (1u << 20)                      // pad_enable
              | (3u << 22)                      // pad_interval: 16 DWORDs
              | (3u << 25);                     // pad_amount: 4 DWORDs (16B)
  g1[0] = (int)w0;
  g1[1] = (int)((tile0 & 0xffffu) << 16);                   // tensor_dim0[15:0] @ bits 63:48
  g1[2] = (int)((tile0 >> 16) | ((tile1 & 0xffffu) << 16)); // tensor_dim0 hi | tensor_dim1 lo
  g1[3] = (int)((tile1 >> 16) | (tile0 << 16));             // tensor_dim1 hi | tile_dim0
  g1[4] = (int)(tile1 & 0xffffu);                           // tile_dim1 | tile_dim2=0
  g1[5] = (int)(unsigned)(stride0 & 0xffffffffu);           // tensor_dim0_stride lo
  g1[6] = (int)(unsigned)((stride0 >> 32) & 0xffffu);       // stride hi | dim1_stride lo = 0
  g1[7] = 0;
  v4i zz4; zz4[0] = 0; zz4[1] = 0; zz4[2] = 0; zz4[3] = 0;
  v8i zz8;
#pragma unroll
  for (int i = 0; i < 8; ++i) zz8[i] = 0;
  __builtin_amdgcn_tensor_load_to_lds(g0, g1, zz4, zz4, zz8, 0);
}
#endif

// ---------------- fp32 -> bf16 convert ----------------
__global__ __launch_bounds__(256) void cvt_bf16(const float* __restrict__ in,
                                                u16* __restrict__ out, int n) {
  int i = blockIdx.x * 256 + threadIdx.x;
  if (i < n) out[i] = f2bf(in[i]);
}

// ---------------- WMMA GEMM: Out = A(MxK) @ W(NxK)^T + bias ----------------
// Block tile 128x128x32, 8 waves (2M x 4N), wave tile 64x32 (4x2 fragments).
// Tiles staged via TDM (tensor_load_to_lds, double-buffered) when available.
// EPI: 0 = f32 out, 1 = bf16 out, 2 = bf16 out + exact GELU
template <int EPI>
__global__ __launch_bounds__(256) void wmma_gemm(
    const u16* __restrict__ A, const u16* __restrict__ W,
    const float* __restrict__ bias,
    float* __restrict__ Of, u16* __restrict__ Ob,
    int K, int Nc) {
  __shared__ u16 As[2][128 * 40];
  __shared__ u16 Ws[2][128 * 40];
  const int tid = threadIdx.x, lane = tid & 31, wave = tid >> 5;
  const int wm = wave & 1, wn = wave >> 1;
  const int m0 = blockIdx.y * 128, n0 = blockIdx.x * 128;
  const int ln16 = lane & 15;

  f32x8 c[4][2];
#pragma unroll
  for (int fi = 0; fi < 4; ++fi)
#pragma unroll
    for (int ni = 0; ni < 2; ++ni)
#pragma unroll
      for (int j = 0; j < 8; ++j) c[fi][ni][j] = 0.f;

#if HAVE_TDM
  int cur = 0;
  if (wave == 0) {  // wave-uniform branch; TDM is a per-wave DMA op
    tdm_load_2d_bf16((unsigned)(size_t)&As[0][0], &A[(size_t)m0 * K], 32u, 128u,
                     (unsigned long long)K);
    tdm_load_2d_bf16((unsigned)(size_t)&Ws[0][0], &W[(size_t)n0 * K], 32u, 128u,
                     (unsigned long long)K);
    __builtin_amdgcn_s_wait_tensorcnt(0);
  }
  __syncthreads();
#endif

  for (int kt = 0; kt < K; kt += 32) {
#if HAVE_TDM
    const int nxt = cur ^ 1;
    if (wave == 0 && kt + 32 < K) {
      tdm_load_2d_bf16((unsigned)(size_t)&As[nxt][0],
                       &A[(size_t)m0 * K + kt + 32], 32u, 128u,
                       (unsigned long long)K);
      tdm_load_2d_bf16((unsigned)(size_t)&Ws[nxt][0],
                       &W[(size_t)n0 * K + kt + 32], 32u, 128u,
                       (unsigned long long)K);
    }
    const u16* Asb = &As[cur][0];
    const u16* Wsb = &Ws[cur][0];
#else
    const int cur = 0;
#pragma unroll
    for (int p = 0; p < 2; ++p) {
      int idx = tid + p * 256;
      int r = idx >> 2, c4 = idx & 3;
      *(uint4*)&As[0][r * 40 + c4 * 8] = *(const uint4*)&A[(size_t)(m0 + r) * K + kt + c4 * 8];
      *(uint4*)&Ws[0][r * 40 + c4 * 8] = *(const uint4*)&W[(size_t)(n0 + r) * K + kt + c4 * 8];
      if (kt + 32 < K) {  // global_prefetch_b8
        __builtin_prefetch(&A[(size_t)(m0 + r) * K + kt + 32 + c4 * 8], 0, 1);
        __builtin_prefetch(&W[(size_t)(n0 + r) * K + kt + 32 + c4 * 8], 0, 1);
      }
    }
    __syncthreads();
    const u16* Asb = &As[0][0];
    const u16* Wsb = &Ws[0][0];
#endif

    bf16x16 af[4], bw[2];
    {
      // A 16x32 layout: lane<16 -> K 0-15 ; lane>=16 -> K 16-31
      const int k0a = (lane < 16) ? 0 : 16;
#pragma unroll
      for (int fi = 0; fi < 4; ++fi) {
        int row = wm * 64 + fi * 16 + ln16;
        FragAB t;
        t.u[0] = *(const uint4*)&Asb[row * 40 + k0a];
        t.u[1] = *(const uint4*)&Asb[row * 40 + k0a + 8];
        af[fi] = t.v;
      }
      // B 32x16 layout: lane holds col=lane%16, K contiguous (0-15 / 16-31)
      const int k0b = (lane < 16) ? 0 : 16;
#pragma unroll
      for (int ni = 0; ni < 2; ++ni) {
        int row = wn * 32 + ni * 16 + ln16;
        FragAB t;
        t.u[0] = *(const uint4*)&Wsb[row * 40 + k0b];
        t.u[1] = *(const uint4*)&Wsb[row * 40 + k0b + 8];
        bw[ni] = t.v;
      }
    }
#pragma unroll
    for (int fi = 0; fi < 4; ++fi)
#pragma unroll
      for (int ni = 0; ni < 2; ++ni)
        c[fi][ni] = __builtin_amdgcn_wmma_f32_16x16x32_bf16(
            false, af[fi], false, bw[ni], (short)0, c[fi][ni], false, false);

#if HAVE_TDM
    if (wave == 0) __builtin_amdgcn_s_wait_tensorcnt(0);
    __syncthreads();
    cur = nxt;
#else
    __syncthreads();
#endif
  }

  const int hi8 = (lane >= 16) ? 8 : 0;
#pragma unroll
  for (int fi = 0; fi < 4; ++fi) {
    const int rbase = m0 + wm * 64 + fi * 16 + hi8;
#pragma unroll
    for (int ni = 0; ni < 2; ++ni) {
      const int col = n0 + wn * 32 + ni * 16 + ln16;
      const float bv = bias[col];
#pragma unroll
      for (int j = 0; j < 8; ++j) {
        float v = c[fi][ni][j] + bv;
        if (EPI == 2) v = 0.5f * v * (1.f + erff(v * 0.70710678118654752f));
        size_t oi = (size_t)(rbase + j) * Nc + col;
        if (EPI == 0) Of[oi] = v;
        else          Ob[oi] = f2bf(v);
      }
    }
  }
}

// ---------------- Flash attention (one wave per (b, h, 16-row q tile)) ------
__global__ __launch_bounds__(256) void attn_flash(
    const u16* __restrict__ Q, const u16* __restrict__ Kb,
    const u16* __restrict__ Vb, const float* __restrict__ rel,
    const unsigned char* __restrict__ nmask, u16* __restrict__ ctx) {
  __shared__ u16 Pl[8][16 * 40];
  const int tid = threadIdx.x, lane = tid & 31, wave = tid >> 5;
  const int gw = blockIdx.x * 8 + wave;
  const int qt = gw & 63;               // N/16 = 64 query tiles
  const int h  = (gw >> 6) % Hh;
  const int b  = gw / (64 * Hh);
  const int ln16 = lane & 15;
  const int hi8  = (lane >= 16) ? 8 : 0;

  f32x8 zero;
#pragma unroll
  for (int j = 0; j < 8; ++j) zero[j] = 0.f;

  // Q A-fragments: 16 rows x DH=64 -> two K=32 fragments
  bf16x16 aq[2];
  {
    const int q = qt * 16 + ln16;
    const u16* qp = Q + ((size_t)(b * Nn + q)) * Dd + h * DHh;
    const int d0 = (lane < 16) ? 0 : 16;
#pragma unroll
    for (int half = 0; half < 2; ++half) {
      FragAB t;
      t.u[0] = *(const uint4*)(qp + half * 32 + d0);
      t.u[1] = *(const uint4*)(qp + half * 32 + d0 + 8);
      aq[half] = t.v;
    }
  }

  f32x8 o[4];
#pragma unroll
  for (int t = 0; t < 4; ++t)
#pragma unroll
    for (int j = 0; j < 8; ++j) o[t][j] = 0.f;
  float m[8], l[8];
#pragma unroll
  for (int j = 0; j < 8; ++j) { m[j] = -INFINITY; l[j] = 0.f; }

  for (int kb = 0; kb < Nn; kb += 32) {
    f32x8 s[2];
#pragma unroll
    for (int ct = 0; ct < 2; ++ct) {
      const int key = kb + ct * 16 + ln16;
      // B fragments of K^T: lane = key column, d contiguous -> 32B loads
      const u16* kp = Kb + ((size_t)(b * Nn + key)) * Dd + h * DHh + ((lane < 16) ? 0 : 16);
      bf16x16 bk0, bk1;
      { FragAB t; t.u[0] = *(const uint4*)(kp);      t.u[1] = *(const uint4*)(kp + 8);  bk0 = t.v; }
      { FragAB t; t.u[0] = *(const uint4*)(kp + 32); t.u[1] = *(const uint4*)(kp + 40); bk1 = t.v; }
      f32x8 acc = __builtin_amdgcn_wmma_f32_16x16x32_bf16(false, aq[0], false, bk0, (short)0, zero, false, false);
      acc = __builtin_amdgcn_wmma_f32_16x16x32_bf16(false, aq[1], false, bk1, (short)0, acc, false, false);
#pragma unroll
      for (int j = 0; j < 8; ++j) acc[j] *= 0.125f;   // 1/sqrt(DH)
      const bool kok = nmask[b * Nn + key] != 0;
      if (h < Rr) {  // typed heads: relation gate (wave-uniform branch)
        const int kcol = kb + ct * 16 + ln16;
#pragma unroll
        for (int j = 0; j < 8; ++j) {
          const int q = qt * 16 + j + hi8;
          const float gate = rel[((size_t)(b * Nn + q) * Nn + kcol) * Rr + h];
          if (gate < 1e-4f) acc[j] = -INFINITY;
        }
      }
      if (!kok) {
#pragma unroll
        for (int j = 0; j < 8; ++j) acc[j] = -INFINITY;
      }
      s[ct] = acc;
    }
    // online softmax: rows live across 16 lanes in the C layout
#pragma unroll
    for (int j = 0; j < 8; ++j) {
      const float rm = redmax16(fmaxf(s[0][j], s[1][j]));
      const float nm = fmaxf(m[j], rm);
      const float alpha = (nm == -INFINITY) ? 1.f : __expf(m[j] - nm);
      const float p0 = (nm == -INFINITY) ? 0.f : __expf(s[0][j] - nm);
      const float p1 = (nm == -INFINITY) ? 0.f : __expf(s[1][j] - nm);
      l[j] = l[j] * alpha + redsum16(p0 + p1);
      m[j] = nm;
#pragma unroll
      for (int t = 0; t < 4; ++t) o[t][j] *= alpha;
      Pl[wave][(j + hi8) * 40 + ln16]      = f2bf(p0);
      Pl[wave][(j + hi8) * 40 + 16 + ln16] = f2bf(p1);
    }
    __syncthreads();
    bf16x16 pf;
    {
      const int d0 = (lane < 16) ? 0 : 16;
      FragAB t;
      t.u[0] = *(const uint4*)&Pl[wave][ln16 * 40 + d0];
      t.u[1] = *(const uint4*)&Pl[wave][ln16 * 40 + d0 + 8];
      pf = t.v;
    }
    __syncthreads();
    const int k0 = (lane < 16) ? 0 : 16;
#pragma unroll
    for (int t = 0; t < 4; ++t) {
      FragW u;
      const u16* vp = Vb + ((size_t)(b * Nn + kb + k0)) * Dd + h * DHh + t * 16 + ln16;
#pragma unroll
      for (int i = 0; i < 8; ++i) {
        const unsigned lo = vp[0];
        const unsigned hi = vp[Dd];
        u.w[i] = lo | (hi << 16);
        vp += 2 * Dd;
      }
      o[t] = __builtin_amdgcn_wmma_f32_16x16x32_bf16(false, pf, false, u.v, (short)0, o[t], false, false);
    }
  }

#pragma unroll
  for (int j = 0; j < 8; ++j) {
    const float inv = (l[j] > 0.f) ? (1.f / l[j]) : 0.f;   // nan_to_num: all-masked row -> 0
    const int q = qt * 16 + j + hi8;
    u16* cp = ctx + ((size_t)(b * Nn + q)) * Dd + h * DHh;
#pragma unroll
    for (int t = 0; t < 4; ++t) cp[t * 16 + ln16] = f2bf(o[t][j] * inv);
  }
}

// ---------------- add + LayerNorm (one block per row of 768) ----------------
__global__ __launch_bounds__(256) void add_ln(
    const float* __restrict__ x, const float* __restrict__ y,
    const float* __restrict__ g, const float* __restrict__ be,
    float* __restrict__ outF, u16* __restrict__ outB) {
  __shared__ float red[256];
  const int row = blockIdx.x, tid = threadIdx.x;
  float v[3];
  float s = 0.f;
#pragma unroll
  for (int i = 0; i < 3; ++i) {
    const int cc = tid + i * 256;
    v[i] = x[(size_t)row * Dd + cc] + y[(size_t)row * Dd + cc];
    s += v[i];
  }
  red[tid] = s; __syncthreads();
  for (int o = 128; o > 0; o >>= 1) { if (tid < o) red[tid] += red[tid + o]; __syncthreads(); }
  const float mean = red[0] * (1.f / Dd);
  __syncthreads();
  float s2 = 0.f;
#pragma unroll
  for (int i = 0; i < 3; ++i) { const float d = v[i] - mean; s2 += d * d; }
  red[tid] = s2; __syncthreads();
  for (int o = 128; o > 0; o >>= 1) { if (tid < o) red[tid] += red[tid + o]; __syncthreads(); }
  const float rstd = rsqrtf(red[0] * (1.f / Dd) + 1e-5f);
#pragma unroll
  for (int i = 0; i < 3; ++i) {
    const int cc = tid + i * 256;
    const float ov = (v[i] - mean) * rstd * g[cc] + be[cc];
    outF[(size_t)row * Dd + cc] = ov;
    if (outB) outB[(size_t)row * Dd + cc] = f2bf(ov);
  }
}

extern "C" void kernel_launch(void* const* d_in, const int* in_sizes, int n_in,
                              void* d_out, int out_size, void* d_ws, size_t ws_size,
                              hipStream_t stream) {
  const float* nodes = (const float*)d_in[0];
  const float* rel   = (const float*)d_in[1];
  const unsigned char* nmask = (const unsigned char*)d_in[2];  // jax bool: 1 byte
  const float* Wq = (const float*)d_in[3];  const float* bq = (const float*)d_in[4];
  const float* Wk = (const float*)d_in[5];  const float* bk = (const float*)d_in[6];
  const float* Wv = (const float*)d_in[7];  const float* bv = (const float*)d_in[8];
  const float* Wo = (const float*)d_in[9];  const float* bo = (const float*)d_in[10];
  const float* ln1g = (const float*)d_in[11]; const float* ln1b = (const float*)d_in[12];
  const float* W1 = (const float*)d_in[13]; const float* b1 = (const float*)d_in[14];
  const float* W2 = (const float*)d_in[15]; const float* b2 = (const float*)d_in[16];
  const float* ln2g = (const float*)d_in[17]; const float* ln2b = (const float*)d_in[18];
  float* out = (float*)d_out;

  char* wsb = (char*)d_ws;
  size_t off = 0;
  auto take = [&](size_t bytes) -> void* {
    off = (off + 255) & ~(size_t)255;
    void* p = wsb + off;
    off += bytes;
    return p;
  };
  u16* Xb  = (u16*)take((size_t)Mrows * Dd * 2);
  u16* Wqb = (u16*)take((size_t)Dd * Dd * 2);
  u16* Wkb = (u16*)take((size_t)Dd * Dd * 2);
  u16* Wvb = (u16*)take((size_t)Dd * Dd * 2);
  u16* Wob = (u16*)take((size_t)Dd * Dd * 2);
  u16* W1b = (u16*)take((size_t)DFf * Dd * 2);
  u16* W2b = (u16*)take((size_t)Dd * DFf * 2);
  u16* Qb  = (u16*)take((size_t)Mrows * Dd * 2);
  u16* Kbw = (u16*)take((size_t)Mrows * Dd * 2);
  u16* Vbw = (u16*)take((size_t)Mrows * Dd * 2);
  u16* CTX = (u16*)take((size_t)Mrows * Dd * 2);
  float* AT = (float*)take((size_t)Mrows * Dd * 4);
  float* H1 = (float*)take((size_t)Mrows * Dd * 4);
  u16* H1b = (u16*)take((size_t)Mrows * Dd * 2);
  u16* Gb  = (u16*)take((size_t)Mrows * DFf * 2);
  float* Ff = (float*)take((size_t)Mrows * Dd * 4);

  const int thr = 256;
  cvt_bf16<<<((size_t)Mrows * Dd + 255) / 256, thr, 0, stream>>>(nodes, Xb, Mrows * Dd);
  cvt_bf16<<<(Dd * Dd + 255) / 256, thr, 0, stream>>>(Wq, Wqb, Dd * Dd);
  cvt_bf16<<<(Dd * Dd + 255) / 256, thr, 0, stream>>>(Wk, Wkb, Dd * Dd);
  cvt_bf16<<<(Dd * Dd + 255) / 256, thr, 0, stream>>>(Wv, Wvb, Dd * Dd);
  cvt_bf16<<<(Dd * Dd + 255) / 256, thr, 0, stream>>>(Wo, Wob, Dd * Dd);
  cvt_bf16<<<(DFf * Dd + 255) / 256, thr, 0, stream>>>(W1, W1b, DFf * Dd);
  cvt_bf16<<<(Dd * DFf + 255) / 256, thr, 0, stream>>>(W2, W2b, Dd * DFf);

  dim3 gD(Dd / 128, Mrows / 128);    // (6, 64)
  dim3 gF(DFf / 128, Mrows / 128);   // (12, 64)

  wmma_gemm<1><<<gD, thr, 0, stream>>>(Xb, Wqb, bq, nullptr, Qb, Dd, Dd);
  wmma_gemm<1><<<gD, thr, 0, stream>>>(Xb, Wkb, bk, nullptr, Kbw, Dd, Dd);
  wmma_gemm<1><<<gD, thr, 0, stream>>>(Xb, Wvb, bv, nullptr, Vbw, Dd, Dd);

  attn_flash<<<(Bdim * Hh * (Nn / 16)) / 8, thr, 0, stream>>>(Qb, Kbw, Vbw, rel, nmask, CTX);

  wmma_gemm<0><<<gD, thr, 0, stream>>>(CTX, Wob, bo, AT, nullptr, Dd, Dd);
  add_ln<<<Mrows, thr, 0, stream>>>(nodes, AT, ln1g, ln1b, H1, H1b);
  wmma_gemm<2><<<gF, thr, 0, stream>>>(H1b, W1b, b1, nullptr, Gb, Dd, DFf);
  wmma_gemm<0><<<gD, thr, 0, stream>>>(Gb, W2b, b2, Ff, nullptr, DFf, Dd);
  add_ln<<<Mrows, thr, 0, stream>>>(H1, Ff, ln2g, ln2b, out, nullptr);
  (void)in_sizes; (void)n_in; (void)out_size; (void)ws_size;
}
